// Classifier_21466246545786
// MI455X (gfx1250) — matrix operand order repaired and checked by reading, not compiled, not run on hardware
//
#include <hip/hip_runtime.h>
#include <hip/hip_bf16.h>

// ---------------------------------------------------------------------------
// PointCNN-style classifier forward pass for MI455X (gfx1250, wave32).
// - GEMM stages: f16-packed operands, v_wmma_f32_16x16x32_f16, dual accums,
//   aligned global_load_b128 fragment loads + global_prefetch_b8.
// - fX stage: LDS tile staged via Tensor Data Mover (tensor_load_to_lds,
//   s_wait_tensorcnt) with a hand-built D# descriptor (ISA cdna5 ch.8).
// ---------------------------------------------------------------------------

typedef __attribute__((ext_vector_type(16))) _Float16 v16h;
typedef __attribute__((ext_vector_type(8)))  _Float16 v8h;
typedef __attribute__((ext_vector_type(8)))  float    v8f;
typedef __attribute__((ext_vector_type(4)))  unsigned int u32x4;
typedef __attribute__((ext_vector_type(8)))  int      i32x8;
typedef __attribute__((ext_vector_type(4)))  int      i32x4;

static inline int cdiv_i(int a, int b) { return (a + b - 1) / b; }
static inline int pad_up(int a, int b) { return cdiv_i(a, b) * b; }

// ---------------------------------------------------------------------------
// TDM: 1-D contiguous tile of `nelem4` floats, global -> LDS.
// D# group0: count=1, lds_addr, global_addr(57b), type=2.
// D# group1: data_size=4B, tensor_dim0=tile_dim0=nelem4, tensor_dim1=1,
//            stride0=nelem4, no pad/iterate/multicast (not in a cluster).
// ---------------------------------------------------------------------------
__device__ __forceinline__ void tdm_load_1d_to_lds(unsigned lds_addr,
                                                   const void* gaddr,
                                                   int nelem4)
{
    unsigned long long ga = (unsigned long long)(uintptr_t)gaddr;
    u32x4 g0;
    g0[0] = 1u;                                          // count=1, user mode
    g0[1] = lds_addr;                                    // LDS byte address
    g0[2] = (unsigned)(ga & 0xFFFFFFFFu);                // global_addr[31:0]
    g0[3] = (unsigned)((ga >> 32) & 0x01FFFFFFu) | 0x80000000u; // [56:32]|type=2
    i32x8 g1;
    g1[0] = 0x00020000;                                  // data_size=2 (4B)
    g1[1] = (nelem4 & 0xFFFF) << 16;                     // tensor_dim0[15:0]
    g1[2] = ((nelem4 >> 16) & 0xFFFF) | (1 << 16);       // dim0[31:16] | dim1=1
    g1[3] = (nelem4 & 0xFFFF) << 16;                     // dim1[31:16]=0 | tile_dim0
    g1[4] = 0;                                           // tile_dim1=0, tile_dim2=0
    g1[5] = nelem4;                                      // tensor_dim0_stride lo
    g1[6] = 0;
    g1[7] = 0;
    i32x4 z4 = {0, 0, 0, 0};
#if defined(__clang_major__) && __clang_major__ >= 23
    i32x8 z8 = {0, 0, 0, 0, 0, 0, 0, 0};
    __builtin_amdgcn_tensor_load_to_lds(g0, g1, z4, z4, z8, 0);
#else
    __builtin_amdgcn_tensor_load_to_lds(g0, g1, z4, z4, 0);
#endif
}

// ---------------------------------------------------------------------------
// Pack A (M x Kd f32, row stride lda) -> f16 row-major, K padded to Kpad.
// ---------------------------------------------------------------------------
__global__ void pack_a_kernel(const float* __restrict__ A, int lda,
                              _Float16* __restrict__ out, int M, int Kd, int Kpad)
{
    int t = blockIdx.x * blockDim.x + threadIdx.x;
    if (t >= M * Kpad) return;
    int row = t / Kpad, k = t % Kpad;
    out[t] = (k < Kd) ? (_Float16)A[(size_t)row * lda + k] : (_Float16)0.0f;
}

// ---------------------------------------------------------------------------
// Pack B (Kd x N f32, row stride ldb) -> f16 COLUMN-major (N16 x Kpad).
// ---------------------------------------------------------------------------
__global__ void pack_bt_kernel(const float* __restrict__ Bw, int ldb,
                               _Float16* __restrict__ out, int N, int Kd,
                               int N16, int Kpad)
{
    int t = blockIdx.x * blockDim.x + threadIdx.x;
    if (t >= N16 * Kpad) return;
    int col = t / Kpad, k = t % Kpad;
    out[t] = (k < Kd && col < N) ? (_Float16)Bw[(size_t)k * ldb + col]
                                 : (_Float16)0.0f;
}

// ---------------------------------------------------------------------------
// WMMA GEMM on packed operands: Y[M,N] = act( Af @ Bt^T + bias ).
// One wave per 16x16 tile; M must be a multiple of 16 (true for all calls).
// ---------------------------------------------------------------------------
__device__ __forceinline__ v16h load_a_frag(const _Float16* __restrict__ arow,
                                            int k0, int g)
{
    v8h lo = *(const v8h*)(arow + k0 + 16 * g);
    v8h hi = *(const v8h*)(arow + k0 + 16 * g + 8);
    return __builtin_shufflevector(lo, hi, 0,1,2,3,4,5,6,7,8,9,10,11,12,13,14,15);
}

__global__ void __launch_bounds__(32)
gemm_wmma_kernel(const _Float16* __restrict__ Af,
                 const _Float16* __restrict__ Bt,
                 const float* __restrict__ bias,
                 float* __restrict__ Y, int ldy,
                 int M, int N, int Kpad, int act)
{
    const int lane = threadIdx.x;      // 0..31
    const int g    = lane >> 4;
    const int mr   = lane & 15;
    const int m0   = blockIdx.y * 16;
    const int n0   = blockIdx.x * 16;

    const _Float16* __restrict__ arow = Af + (size_t)(m0 + mr) * Kpad;
    const _Float16* __restrict__ brow = Bt + (size_t)(n0 + mr) * Kpad;

    v8f acc0 = {0.f,0.f,0.f,0.f,0.f,0.f,0.f,0.f};
    v8f acc1 = {0.f,0.f,0.f,0.f,0.f,0.f,0.f,0.f};

    int k0 = 0;
    for (; k0 + 64 <= Kpad; k0 += 64) {
        __builtin_prefetch(arow + k0 + 64, 0, 1);
        v16h a0 = load_a_frag(arow, k0, g);
        v16h b0 = *(const v16h*)(brow + k0 + 16 * g);
        v16h a1 = load_a_frag(arow, k0 + 32, g);
        v16h b1 = *(const v16h*)(brow + k0 + 32 + 16 * g);
        acc0 = __builtin_amdgcn_wmma_f32_16x16x32_f16(
                   false, a0, false, b0, (short)0, acc0, false, false);
        acc1 = __builtin_amdgcn_wmma_f32_16x16x32_f16(
                   false, a1, false, b1, (short)0, acc1, false, false);
    }
    if (k0 < Kpad) {
        v16h a0 = load_a_frag(arow, k0, g);
        v16h b0 = *(const v16h*)(brow + k0 + 16 * g);
        acc0 = __builtin_amdgcn_wmma_f32_16x16x32_f16(
                   false, a0, false, b0, (short)0, acc0, false, false);
    }

    const int col = n0 + mr;
    if (col < N) {
        const float bv = bias ? bias[col] : 0.0f;
        #pragma unroll
        for (int r = 0; r < 8; ++r) {
            const int row = m0 + r + 8 * g;   // always < M (M % 16 == 0)
            float v = acc0[r] + acc1[r] + bv;
            if (act) v = fmaxf(v, 0.0f);
            Y[(size_t)row * ldy + col] = v;
        }
    }
}

// ---------------------------------------------------------------------------
// Deterministic subsample: rep[b,j,:] = pts[b, (j*997)%N, :]
// ---------------------------------------------------------------------------
__global__ void subsample_kernel(const float* __restrict__ pts,
                                 float* __restrict__ rep, int B, int N, int P)
{
    int t = blockIdx.x * blockDim.x + threadIdx.x;
    if (t >= B * P) return;
    int b = t / P, j = t % P;
    int s = (j * 997) % N;
    const float* src = pts + ((size_t)b * N + s) * 3;
    float* dst = rep + (size_t)t * 3;
    dst[0] = src[0]; dst[1] = src[1]; dst[2] = src[2];
}

// ---------------------------------------------------------------------------
// KNN: per query thread, keep K*D+1 smallest d2, emit idx[1::D].
// ---------------------------------------------------------------------------
#define KNN_MAX 73  // max K*D+1 (layer 4: 12*6+1)

__global__ void knn_kernel(const float* __restrict__ rep,
                           const float* __restrict__ pts,
                           int* __restrict__ idx,
                           int B, int P, int N, int K, int D)
{
    int t = blockIdx.x * blockDim.x + threadIdx.x;
    if (t >= B * P) return;
    int b = t / P, p = t % P;
    const float* rp = rep + ((size_t)b * P + p) * 3;
    const float rx = rp[0], ry = rp[1], rz = rp[2];
    const float* pb = pts + (size_t)b * N * 3;

    const int Mq = K * D + 1;
    float dist[KNN_MAX + 1];
    int   ind[KNN_MAX + 1];
    int cnt = 0;

    for (int n = 0; n < N; ++n) {
        float dx = pb[n * 3 + 0] - rx;
        float dy = pb[n * 3 + 1] - ry;
        float dz = pb[n * 3 + 2] - rz;
        float d2 = dx * dx + dy * dy + dz * dz;
        if (cnt == Mq && d2 >= dist[Mq - 1]) continue;
        int pos = (cnt < Mq) ? cnt : Mq - 1;
        while (pos > 0 && dist[pos - 1] > d2) {
            dist[pos] = dist[pos - 1];
            ind[pos]  = ind[pos - 1];
            --pos;
        }
        dist[pos] = d2;
        ind[pos]  = n;
        if (cnt < Mq) ++cnt;
    }
    int* op = idx + (size_t)t * K;
    for (int kk = 0; kk < K; ++kk) op[kk] = ind[1 + kk * D];
}

// ---------------------------------------------------------------------------
// Gather: loc[bpk,:] = pts[idx] - rep ; fcat[bpk, cmid:C] = f[idx]
// ---------------------------------------------------------------------------
__global__ void gather_kernel(const float* __restrict__ pts,
                              const float* __restrict__ rep,
                              const float* __restrict__ f,
                              const int* __restrict__ idx,
                              float* __restrict__ loc,
                              float* __restrict__ fcat,
                              int B, int P, int N, int K,
                              int cinx, int cmid, int C)
{
    int t = blockIdx.x * blockDim.x + threadIdx.x;
    if (t >= B * P * K) return;
    int b  = t / (P * K);
    int r  = t % (P * K);
    int p  = r / K;
    int nb = idx[t];
    const float* pn = pts + ((size_t)b * N + nb) * 3;
    const float* rp = rep + ((size_t)b * P + p) * 3;
    float* lo = loc + (size_t)t * 3;
    lo[0] = pn[0] - rp[0];
    lo[1] = pn[1] - rp[1];
    lo[2] = pn[2] - rp[2];
    const float* fr = f + ((size_t)b * N + nb) * cinx;
    float* fc = fcat + (size_t)t * C + cmid;
    for (int c = 0; c < cinx; ++c) fc[c] = fr[c];
}

// ---------------------------------------------------------------------------
// Transpose xw (K^2, 3K) -> xwT (3K, K^2)
// ---------------------------------------------------------------------------
__global__ void transpose_kernel(const float* __restrict__ xw,
                                 float* __restrict__ xwT, int KK, int K3)
{
    int t = blockIdx.x * blockDim.x + threadIdx.x;
    if (t >= KK * K3) return;
    int kk = t / KK, o = t % KK;
    xwT[(size_t)kk * KK + o] = xw[(size_t)o * K3 + kk];
}

// ---------------------------------------------------------------------------
// fX in-place: stage fcat tile (K x C floats) into LDS via the Tensor Data
// Mover (wave 0 issues, s_wait_tensorcnt, then block barrier), then
// fcat[bp,i,c] = sum_j X[bp,i,j] * tile[j,c]. One workgroup per point.
// ---------------------------------------------------------------------------
__global__ void fx_inplace_kernel(float* __restrict__ fcat,
                                  const float* __restrict__ X,
                                  int K, int C)
{
    extern __shared__ float tile[];  // K*C floats
    const int bp = blockIdx.x;
    float* base = fcat + (size_t)bp * K * C;

    if (threadIdx.x < 32) {  // one wave issues the TDM op (EXEC-ignored)
        tdm_load_1d_to_lds((unsigned)(uintptr_t)tile, base, K * C);
        __builtin_amdgcn_s_wait_tensorcnt(0);
    }
    __syncthreads();

    const float* Xp = X + (size_t)bp * K * K;
    for (int o = threadIdx.x; o < K * C; o += blockDim.x) {
        int i = o / C, c = o % C;
        float s = 0.0f;
        for (int j = 0; j < K; ++j) s += Xp[i * K + j] * tile[j * C + c];
        base[o] = s;
    }
}

// ---------------------------------------------------------------------------
// Depthwise: dwc[bp, c*dm+m] = sum_k fX[bp,k,c]*dw[c,m,k] + db[c,m]
// ---------------------------------------------------------------------------
__global__ void dwc_kernel(const float* __restrict__ fX,
                           const float* __restrict__ dw,
                           const float* __restrict__ db,
                           float* __restrict__ out,
                           int BP, int K, int C, int dm)
{
    int t = blockIdx.x * blockDim.x + threadIdx.x;
    if (t >= BP * C * dm) return;
    int bp = t / (C * dm);
    int r  = t % (C * dm);
    int c  = r / dm, m = r % dm;
    const float* fx = fX + (size_t)bp * K * C + c;
    const float* dp = dw + ((size_t)c * dm + m) * K;
    float s = db[r];
    for (int k = 0; k < K; ++k) s += fx[(size_t)k * C] * dp[k];
    out[t] = s;
}

// ---------------------------------------------------------------------------
// Mean over points: out[b, cls] = mean_p logits[b, p, cls]
// ---------------------------------------------------------------------------
__global__ void mean_kernel(const float* __restrict__ logits,
                            float* __restrict__ out, int B, int Np, int NC)
{
    int t = blockIdx.x * blockDim.x + threadIdx.x;
    if (t >= B * NC) return;
    int b = t / NC, c = t % NC;
    float s = 0.0f;
    for (int p = 0; p < Np; ++p) s += logits[((size_t)b * Np + p) * NC + c];
    out[t] = s / (float)Np;
}

// ---------------------------------------------------------------------------
// Host orchestration
// ---------------------------------------------------------------------------
struct LayerC { int cin, cout, K, D, P, cinx, cmid, C, dm; };
static const LayerC LC[5] = {
    {  3,  32,  8, 1,  -1, 16,  8,  24, 4},
    { 32,  64,  8, 2,  -1, 32, 16,  48, 2},
    { 64,  96,  8, 4,  -1, 48, 24,  72, 2},
    { 96, 128, 12, 4, 120, 64, 32,  96, 2},
    {128, 160, 12, 6, 120, 80, 40, 120, 2},
};

struct GemmScratch { _Float16* Af; _Float16* Bt; };

static void launch_gemm(const float* A, int lda, const float* Bw, int ldb,
                        const float* bias, float* Y, int ldy,
                        int M, int N, int Kd, int act,
                        const GemmScratch& gs, hipStream_t s)
{
    const int Kpad = pad_up(Kd, 32);
    const int N16  = pad_up(N, 16);
    pack_a_kernel<<<cdiv_i(M * Kpad, 256), 256, 0, s>>>(A, lda, gs.Af, M, Kd, Kpad);
    pack_bt_kernel<<<cdiv_i(N16 * Kpad, 256), 256, 0, s>>>(Bw, ldb, gs.Bt, N, Kd, N16, Kpad);
    dim3 grid(N16 / 16, M / 16);
    gemm_wmma_kernel<<<grid, 32, 0, s>>>(gs.Af, gs.Bt, bias, Y, ldy, M, N, Kpad, act);
}

extern "C" void kernel_launch(void* const* d_in, const int* in_sizes, int n_in,
                              void* d_out, int out_size, void* d_ws, size_t ws_size,
                              hipStream_t stream)
{
    (void)in_sizes; (void)n_in; (void)out_size; (void)ws_size;
    const int B = 16;
    int Ncur = 2048;
    const float* pts_cur = (const float*)d_in[0];
    const float* fts_cur = (const float*)d_in[1];

    auto IN = [&](int i) -> const float* { return (const float*)d_in[i]; };
    const float* h0b = IN(2); const float* h0w = IN(3);
    const float* h1b = IN(4); const float* h1w = IN(5);
    const float* h2b = IN(6); const float* h2w = IN(7);

    // ---- workspace carving (bump allocator, 256B aligned) ----
    char* wp = (char*)d_ws;
    auto alloc = [&](size_t bytes) -> void* {
        void* p = (void*)wp;
        wp += (bytes + 255) & ~(size_t)255;
        return p;
    };
    float* REP  = (float*)alloc((size_t)16 * 120 * 3 * 4);
    float* F    = (float*)alloc((size_t)32768 * 64 * 4);
    int*   IDX  = (int*)  alloc((size_t)262144 * 4);
    float* LOC  = (float*)alloc((size_t)262144 * 3 * 4);
    float* FCAT = (float*)alloc((size_t)262144 * 72 * 4);
    float* T1   = (float*)alloc((size_t)262144 * 24 * 4);
    float* XA   = (float*)alloc((size_t)32768 * 64 * 4);
    float* XB   = (float*)alloc((size_t)32768 * 64 * 4);
    float* DWC  = (float*)alloc((size_t)32768 * 144 * 4);
    float* FT0  = (float*)alloc((size_t)32768 * 96 * 4);
    float* FT1  = (float*)alloc((size_t)32768 * 96 * 4);
    float* XWT  = (float*)alloc((size_t)36 * 144 * 4);
    GemmScratch gs;
    gs.Af = (_Float16*)alloc((size_t)9 * 1024 * 1024 * 2);
    gs.Bt = (_Float16*)alloc((size_t)64 * 1024 * 2);

    int flip = 0;
    for (int i = 0; i < 5; ++i) {
        const LayerC& L = LC[i];
        const int base = 8 + i * 15;
        const float* d1b  = IN(base + 0),  *d1w  = IN(base + 1);
        const float* d2b  = IN(base + 2),  *d2w  = IN(base + 3);
        const float* db   = IN(base + 4);
        const float* dnb  = IN(base + 5),  *dnw  = IN(base + 6);
        const float* dw   = IN(base + 7);
        const float* pw   = IN(base + 8);
        const float* xb   = IN(base + 9);
        const float* xd1b = IN(base + 10), *xd1w = IN(base + 11);
        const float* xd2b = IN(base + 12), *xd2w = IN(base + 13);
        const float* xw   = IN(base + 14);

        const float* rep;
        int Pn;
        if (L.P > 0 && L.P < Ncur) {
            subsample_kernel<<<cdiv_i(B * L.P, 256), 256, 0, stream>>>(
                pts_cur, REP, B, Ncur, L.P);
            rep = REP; Pn = L.P;
        } else {
            rep = pts_cur; Pn = Ncur;
        }

        launch_gemm(fts_cur, L.cin, dnw, L.cinx, dnb, F, L.cinx,
                    B * Ncur, L.cinx, L.cin, 1, gs, stream);

        knn_kernel<<<cdiv_i(B * Pn, 256), 256, 0, stream>>>(
            rep, pts_cur, IDX, B, Pn, Ncur, L.K, L.D);

        const int BP = B * Pn, BPK = BP * L.K;
        gather_kernel<<<cdiv_i(BPK, 256), 256, 0, stream>>>(
            pts_cur, rep, F, IDX, LOC, FCAT, B, Pn, Ncur, L.K, L.cinx, L.cmid, L.C);

        launch_gemm(LOC, 3, d1w, L.cmid, d1b, T1, L.cmid, BPK, L.cmid, 3, 1, gs, stream);
        launch_gemm(T1, L.cmid, d2w, L.cmid, d2b, FCAT, L.C, BPK, L.cmid, L.cmid, 1, gs, stream);

        const int KK = L.K * L.K, K3 = L.K * 3;
        transpose_kernel<<<cdiv_i(KK * K3, 256), 256, 0, stream>>>(xw, XWT, KK, K3);
        launch_gemm(LOC, K3, XWT, KK, xb, XA, KK, BP, KK, K3, 1, gs, stream);
        launch_gemm(XA, KK, xd1w, KK, xd1b, XB, KK, BP, KK, KK, 1, gs, stream);
        launch_gemm(XB, KK, xd2w, KK, xd2b, XA, KK, BP, KK, KK, 0, gs, stream);

        fx_inplace_kernel<<<BP, 128, L.K * L.C * 4, stream>>>(FCAT, XA, L.K, L.C);

        const int Cdm = L.C * L.dm;
        dwc_kernel<<<cdiv_i(BP * Cdm, 256), 256, 0, stream>>>(
            FCAT, dw, db, DWC, BP, L.K, L.C, L.dm);

        float* OUT = flip ? FT1 : FT0;
        launch_gemm(DWC, Cdm, pw, L.cout, nullptr, OUT, L.cout,
                    BP, L.cout, Cdm, 1, gs, stream);

        pts_cur = rep;
        fts_cur = OUT;
        Ncur = Pn;
        flip ^= 1;
    }

    float* H1  = F;
    float* H2  = T1;
    float* LOG = XA;
    const int Mh = B * Ncur;  // 16 * 120 = 1920
    launch_gemm(fts_cur, 160, h0w, 128, h0b, H1, 128, Mh, 128, 160, 1, gs, stream);
    launch_gemm(H1, 128, h1w, 64, h1b, H2, 64, Mh, 64, 128, 1, gs, stream);
    launch_gemm(H2, 64, h2w, 10, h2b, LOG, 10, Mh, 10, 64, 0, gs, stream);
    mean_kernel<<<cdiv_i(B * 10, 64), 64, 0, stream>>>(LOG, (float*)d_out, B, Ncur, 10);
}
